// Gcn_57715770524247
// MI455X (gfx1250) — compile-verified, matrix operand order checked by reading, not run on hardware
//
#include <hip/hip_runtime.h>

#define NN 100000   // nodes
#define NE 1000000  // edges
#define NG 512      // graphs
#define NC 10       // classes
#define F  64       // hidden width

typedef __attribute__((ext_vector_type(16))) __bf16 v16bf;
typedef __attribute__((ext_vector_type(8)))  float  v8f;

static inline int cdiv(int a, int b) { return (a + b - 1) / b; }

// ---------------- elementwise / setup kernels ----------------

__global__ void k_fill(float* __restrict__ p, float v, int n) {
  int i = blockIdx.x * blockDim.x + threadIdx.x;
  if (i < n) p[i] = v;
}

__global__ void k_deg_accum(const int* __restrict__ col, float* __restrict__ deg, int e) {
  int i = blockIdx.x * blockDim.x + threadIdx.x;
  if (i < e) atomicAdd(&deg[col[i]], 1.0f);
}

__global__ void k_dinv(float* __restrict__ d, int n) {
  int i = blockIdx.x * blockDim.x + threadIdx.x;
  if (i < n) d[i] = rsqrtf(d[i]);   // deg >= 1 always (self loop)
}

// pack a 64x64 row-major f32 weight W[k][n] into WMMA B-fragment layout:
// 8 tiles (kk in 0..1, nt in 0..3), each tile = 32 lanes x 16 bf16 contiguous.
// lane L, elem i: k = kk*32 + (i<8?0:16) + (L>=16?8:0) + (i&7); n = nt*16 + (L&15)
__global__ void k_pack_w(const float* __restrict__ W, __bf16* __restrict__ pk) {
  int idx = blockIdx.x * blockDim.x + threadIdx.x;
  if (idx >= 8 * 32 * 16) return;
  int t = idx >> 9;          // tile 0..7
  int rem = idx & 511;
  int L = rem >> 4;          // lane 0..31
  int i = rem & 15;          // elem 0..15
  int kk = t >> 2, nt = t & 3;
  int k = kk * 32 + ((i & 8) ? 16 : 0) + ((L >= 16) ? 8 : 0) + (i & 7);
  int n = nt * 16 + (L & 15);
  pk[idx] = (__bf16)W[k * F + n];
}

// layer 0 linear: concat(pos, x) @ gW0  (Fin=3, trivial)
__global__ void k_feat0(const float* __restrict__ x, const float* __restrict__ pos,
                        const float* __restrict__ W, float* __restrict__ lin, int n) {
  int i = blockIdx.x * blockDim.x + threadIdx.x;
  if (i >= n * F) return;
  int node = i >> 6, f = i & 63;
  lin[i] = pos[node * 2 + 0] * W[0 * F + f]
         + pos[node * 2 + 1] * W[1 * F + f]
         + x[node]           * W[2 * F + f];
}

// edge aggregation: out[col] += lin[row] * dinv[row]*dinv[col]  (float4 per thread)
__global__ void k_agg(const int* __restrict__ row, const int* __restrict__ col,
                      const float* __restrict__ dinv, const float* __restrict__ lin,
                      float* __restrict__ out, int e) {
  int i = blockIdx.x * blockDim.x + threadIdx.x;
  if (i >= e * 16) return;
  int ed = i >> 4, fg = i & 15;
  int r = row[ed], c = col[ed];
  float nrm = dinv[r] * dinv[c];
  float4 v = ((const float4*)lin)[(size_t)r * 16 + fg];
  float* o = out + (size_t)c * F + fg * 4;
  atomicAdd(o + 0, v.x * nrm);
  atomicAdd(o + 1, v.y * nrm);
  atomicAdd(o + 2, v.z * nrm);
  atomicAdd(o + 3, v.w * nrm);
}

// h = relu(agg + lin*dinv^2 (self loop) + bias)   (in place on agg)
__global__ void k_sbr(float* __restrict__ agg, const float* __restrict__ lin,
                      const float* __restrict__ dinv, const float* __restrict__ b, int n) {
  int i = blockIdx.x * blockDim.x + threadIdx.x;
  if (i >= n * F) return;
  int node = i >> 6, f = i & 63;
  float di = dinv[node];
  float v = agg[i] + lin[i] * di * di + b[f];
  agg[i] = v > 0.f ? v : 0.f;
}

// mean-pool accumulate
__global__ void k_pool(const float* __restrict__ h, const int* __restrict__ batch,
                       float* __restrict__ gsum, float* __restrict__ gcnt, int n) {
  int i = blockIdx.x * blockDim.x + threadIdx.x;
  if (i >= n * 16) return;
  int node = i >> 4, fg = i & 15;
  int g = batch[node];
  float4 v = ((const float4*)h)[(size_t)node * 16 + fg];
  float* o = gsum + (size_t)g * F + fg * 4;
  atomicAdd(o + 0, v.x);
  atomicAdd(o + 1, v.y);
  atomicAdd(o + 2, v.z);
  atomicAdd(o + 3, v.w);
  if (fg == 0) atomicAdd(&gcnt[g], 1.0f);
}

__global__ void k_pool_div(const float* __restrict__ gsum, const float* __restrict__ gcnt,
                           float* __restrict__ pooled, int g) {
  int i = blockIdx.x * blockDim.x + threadIdx.x;
  if (i >= g * F) return;
  float c = gcnt[i >> 6];
  pooled[i] = gsum[i] / (c > 1.0f ? c : 1.0f);
}

__global__ void k_logits(const float* __restrict__ h, const float* __restrict__ W,
                         const float* __restrict__ b, float* __restrict__ out) {
  int i = blockIdx.x * blockDim.x + threadIdx.x;
  if (i >= NG * NC) return;
  int g = i / NC, c = i % NC;
  float acc = b[c];
#pragma unroll 8
  for (int f = 0; f < F; ++f) acc += h[(size_t)g * F + f] * W[f * NC + c];
  out[i] = acc;
}

// ---------------- WMMA dense transform: out[nrows,64] = in[nrows,64] @ W ----------------
// one wave per 16-row tile; A in bf16 from row-major f32, B prepacked bf16 fragments.
__global__ void k_wmma_linear(const float* __restrict__ in, const __bf16* __restrict__ wp,
                              const float* __restrict__ bias, float* __restrict__ out,
                              int nrows, int fuse_relu) {
  int wave = threadIdx.x >> 5;
  int lane = threadIdx.x & 31;
  int tile = blockIdx.x * 8 + wave;           // blockDim.x == 256 -> 8 waves
  int row0 = tile * 16;
  if (row0 >= nrows) return;                  // wave-uniform: EXEC stays all-1s for WMMA
  int m  = lane & 15;
  int hi = lane >> 4;                         // 0: lanes 0-15, 1: lanes 16-31
  const float* rowp = in + (size_t)(row0 + m) * F;

  v8f acc0 = {}, acc1 = {}, acc2 = {}, acc3 = {};
#pragma unroll
  for (int kk = 0; kk < 2; ++kk) {
    // A fragment: 16x32 bf16 (ISA layout: this half-wave's K chunks are
    // [kbase..kbase+7] and [kbase+16..kbase+23])
    v16bf a;
    int kbase = kk * 32 + hi * 8;
#pragma unroll
    for (int i = 0; i < 8; ++i) {
      a[i]     = (__bf16)rowp[kbase + i];
      a[i + 8] = (__bf16)rowp[kbase + 16 + i];
    }
    const __bf16* wt = wp + (size_t)(kk * 4) * 512 + lane * 16;
    v16bf b0 = *(const v16bf*)(wt + 0 * 512);
    v16bf b1 = *(const v16bf*)(wt + 1 * 512);
    v16bf b2 = *(const v16bf*)(wt + 2 * 512);
    v16bf b3 = *(const v16bf*)(wt + 3 * 512);
    acc0 = __builtin_amdgcn_wmma_f32_16x16x32_bf16(false, a, false, b0, (short)0, acc0, false, false);
    acc1 = __builtin_amdgcn_wmma_f32_16x16x32_bf16(false, a, false, b1, (short)0, acc1, false, false);
    acc2 = __builtin_amdgcn_wmma_f32_16x16x32_bf16(false, a, false, b2, (short)0, acc2, false, false);
    acc3 = __builtin_amdgcn_wmma_f32_16x16x32_bf16(false, a, false, b3, (short)0, acc3, false, false);
  }

  float b0v = 0.f, b1v = 0.f, b2v = 0.f, b3v = 0.f;
  if (bias) { b0v = bias[m]; b1v = bias[16 + m]; b2v = bias[32 + m]; b3v = bias[48 + m]; }
#pragma unroll
  for (int r = 0; r < 8; ++r) {
    int mm = r + hi * 8;                      // C/D layout: VGPR r -> M = r (+8 for hi lanes)
    float* op = out + (size_t)(row0 + mm) * F + m;
    float v0 = acc0[r] + b0v, v1 = acc1[r] + b1v, v2 = acc2[r] + b2v, v3 = acc3[r] + b3v;
    if (fuse_relu) {
      v0 = v0 > 0.f ? v0 : 0.f; v1 = v1 > 0.f ? v1 : 0.f;
      v2 = v2 > 0.f ? v2 : 0.f; v3 = v3 > 0.f ? v3 : 0.f;
    }
    op[0] = v0; op[16] = v1; op[32] = v2; op[48] = v3;
  }
}

// ---------------- host launcher ----------------

extern "C" void kernel_launch(void* const* d_in, const int* in_sizes, int n_in,
                              void* d_out, int out_size, void* d_ws, size_t ws_size,
                              hipStream_t stream) {
  const float* x    = (const float*)d_in[0];
  const float* pos  = (const float*)d_in[1];
  const int*   ei   = (const int*)d_in[2];
  const int*   batch= (const int*)d_in[3];
  const float* gW0  = (const float*)d_in[4];
  const float* gb0  = (const float*)d_in[5];
  const float* gW1  = (const float*)d_in[6];
  const float* gb1  = (const float*)d_in[7];
  const float* gW2  = (const float*)d_in[8];
  const float* gb2  = (const float*)d_in[9];
  const float* lW0  = (const float*)d_in[10];
  const float* lb0  = (const float*)d_in[11];
  const float* lW1  = (const float*)d_in[12];
  const float* lb1  = (const float*)d_in[13];
  const int* rowp = ei;
  const int* colp = ei + NE;

  // workspace carve-up (aligned to 256B)
  char* w = (char*)d_ws;
  auto take = [&](size_t bytes) { char* p = w; w += (bytes + 255) & ~(size_t)255; return p; };
  float*  dinv   = (float*)take((size_t)NN * 4);
  float*  bufA   = (float*)take((size_t)NN * F * 4);
  float*  bufB   = (float*)take((size_t)NN * F * 4);
  __bf16* pk1    = (__bf16*)take(8 * 512 * 2);
  __bf16* pk2    = (__bf16*)take(8 * 512 * 2);
  __bf16* pkL    = (__bf16*)take(8 * 512 * 2);
  float*  gsum   = (float*)take((size_t)NG * F * 4);
  float*  gcnt   = (float*)take((size_t)NG * 4);
  float*  pooled = (float*)take((size_t)NG * F * 4);
  float*  mlp1   = (float*)take((size_t)NG * F * 4);

  const int B = 256;
  const int nNF = NN * F;

  // --- degrees / normalization ---
  k_fill<<<cdiv(NN, B), B, 0, stream>>>(dinv, 1.0f, NN);           // self loop
  k_deg_accum<<<cdiv(NE, B), B, 0, stream>>>(colp, dinv, NE);
  k_dinv<<<cdiv(NN, B), B, 0, stream>>>(dinv, NN);

  // --- pack weights for WMMA ---
  k_pack_w<<<cdiv(4096, B), B, 0, stream>>>(gW1, pk1);
  k_pack_w<<<cdiv(4096, B), B, 0, stream>>>(gW2, pk2);
  k_pack_w<<<cdiv(4096, B), B, 0, stream>>>(lW0, pkL);

  // --- layer 0: lin0 = concat(pos,x) @ gW0 -> bufB; aggregate -> bufA; fuse self+bias+relu
  k_feat0<<<cdiv(nNF, B), B, 0, stream>>>(x, pos, gW0, bufB, NN);
  k_fill<<<cdiv(nNF, B), B, 0, stream>>>(bufA, 0.0f, nNF);
  k_agg<<<cdiv(NE * 16, B), B, 0, stream>>>(rowp, colp, dinv, bufB, bufA, NE);
  k_sbr<<<cdiv(nNF, B), B, 0, stream>>>(bufA, bufB, dinv, gb0, NN);

  // --- layers 1 and 2: WMMA linear then aggregate ---
  const int tilesN = NN / 16;                  // 6250
  const int gridW  = cdiv(tilesN, 8);
  k_wmma_linear<<<gridW, B, 0, stream>>>(bufA, pk1, nullptr, bufB, NN, 0);
  k_fill<<<cdiv(nNF, B), B, 0, stream>>>(bufA, 0.0f, nNF);
  k_agg<<<cdiv(NE * 16, B), B, 0, stream>>>(rowp, colp, dinv, bufB, bufA, NE);
  k_sbr<<<cdiv(nNF, B), B, 0, stream>>>(bufA, bufB, dinv, gb1, NN);

  k_wmma_linear<<<gridW, B, 0, stream>>>(bufA, pk2, nullptr, bufB, NN, 0);
  k_fill<<<cdiv(nNF, B), B, 0, stream>>>(bufA, 0.0f, nNF);
  k_agg<<<cdiv(NE * 16, B), B, 0, stream>>>(rowp, colp, dinv, bufB, bufA, NE);
  k_sbr<<<cdiv(nNF, B), B, 0, stream>>>(bufA, bufB, dinv, gb2, NN);

  // --- global mean pool ---
  k_fill<<<cdiv(NG * F, B), B, 0, stream>>>(gsum, 0.0f, NG * F);
  k_fill<<<cdiv(NG, B), B, 0, stream>>>(gcnt, 0.0f, NG);
  k_pool<<<cdiv(NN * 16, B), B, 0, stream>>>(bufA, batch, gsum, gcnt, NN);
  k_pool_div<<<cdiv(NG * F, B), B, 0, stream>>>(gsum, gcnt, pooled, NG);

  // --- MLP head: relu(pooled @ lW0 + lb0) via WMMA, then 64->10 logits ---
  k_wmma_linear<<<cdiv(NG / 16, 8), B, 0, stream>>>(pooled, pkL, lb0, mlp1, NG, 1);
  k_logits<<<cdiv(NG * NC, B), B, 0, stream>>>(mlp1, lW1, lb1, (float*)d_out);
}
